// VQ_38671885534099
// MI455X (gfx1250) — matrix operand order repaired and checked by reading, not compiled, not run on hardware
//
#include <hip/hip_runtime.h>

// ---------------------------------------------------------------------------
// Angular VQ for MI455X (gfx1250, wave32, WMMA).
//   x:        [32, 64, 64, 64] f32   (B, C, H, W)
//   codebook: [512, 64]        f32   (K codes, C)
// Outputs (concatenated in d_out, f32):
//   codes:   [32, 64, 64, 64]  (B, C, H, W)  = normalized codebook rows gathered
//   indices: [32, 64, 64]      (B, H, W)     = argmin index as float
//
// Core: normalized dot-product GEMM [131072 x 64] @ [64 x 512] done with
// v_wmma_f32_16x16x32_f16 (f16 in, f32 accumulate), fused row-wise argmax.
// ---------------------------------------------------------------------------

typedef __attribute__((ext_vector_type(16))) _Float16 v16h;
typedef __attribute__((ext_vector_type(8)))  float    v8f;

#define BDIM   32
#define CDIM   64
#define HDIM   64
#define WDIM   64
#define KCODES 512
#define HW     (HDIM * WDIM)          // 4096
#define NTILES (KCODES / 16)          // 32 code tiles
// Workspace layout:
//   [0, 65536)          : f16 codebook in WMMA B layout: [tile][chunk][lane][16]
//   [65536, 65536+128K) : f32 normalized codebook, row-major [512][64]
#define WS_BMAT_BYTES (NTILES * 2 * 32 * 16 * sizeof(_Float16))

// ---------------------------------------------------------------------------
// Prep: normalize codebook; emit f32 row-major copy + f16 WMMA-B-layout copy.
// B layout (16-bit B 32x16, per ISA 7.12.2 / sparse-B analogy):
//   lane L holds column n = L&15; element e holds K = 8*(L>>4) + e for e<8, K = 16 + 8*(L>>4) + (e-8) for e>=8.
// One block of 512 threads; thread k owns codebook row k.
// ---------------------------------------------------------------------------
__global__ __launch_bounds__(512) void vq_prep_kernel(
    const float* __restrict__ cb, float* __restrict__ cbn,
    _Float16* __restrict__ bmat) {
  int k = threadIdx.x;                 // code row 0..511
  float v[CDIM];
  float s = 0.0f;
#pragma unroll
  for (int c = 0; c < CDIM; ++c) {
    float t = cb[k * CDIM + c];
    v[c] = t;
    s += t * t;
  }
  float inv = 1.0f / fmaxf(sqrtf(s), 1e-12f);
#pragma unroll
  for (int c = 0; c < CDIM; ++c) {
    v[c] *= inv;
    cbn[k * CDIM + c] = v[c];
  }
  int T = k >> 4;                      // code tile
  int n = k & 15;                      // column within tile
#pragma unroll
  for (int q = 0; q < 2; ++q) {        // K-chunk: channels 32q..32q+31
#pragma unroll
    for (int half = 0; half < 2; ++half) {
      int L = n + half * 16;
      _Float16* p = bmat + (((T * 2 + q) * 32 + L) * 16);
#pragma unroll
      for (int e = 0; e < 16; ++e)
        p[e] = (_Float16)v[32 * q + 8 * half + e + (e & 8)];
    }
  }
}

// ---------------------------------------------------------------------------
// Main: one wave handles 16 pixels sharing (b,h), w = w0..w0+15 (A rows),
// against all 512 codes (32 B tiles), fused argmax + output scatter.
// ---------------------------------------------------------------------------
__global__ __launch_bounds__(256) void vq_main_kernel(
    const float* __restrict__ x, const float* __restrict__ cbn,
    const _Float16* __restrict__ bmat, float* __restrict__ codes_out,
    float* __restrict__ idx_out) {
  const int lane = threadIdx.x & 31;
  const int wave = blockIdx.x * (blockDim.x >> 5) + (threadIdx.x >> 5);
  // wave in [0, 8192): (b, h, wtile)
  const int wt = wave & 3;
  const int h  = (wave >> 2) & (HDIM - 1);
  const int b  = wave >> 8;
  const int w0 = wt * 16;
  const int m  = lane & 15;            // A-matrix row (pixel w0+m)

  // ---- Load this lane's A-slot channels (ISA 16-bit A 16x32 layout):
  //   lanes 0-15 : elems 0..7 -> c 0..7,   elems 8..15 -> c 16..23 (chunk0)
  //   lanes 16-31: elems 0..7 -> c 8..15,  elems 8..15 -> c 24..31
  //   chunk1 = same pattern + 32.
  const int cbase = (lane < 16) ? 0 : 8;
  const float* xp = x + (size_t)b * CDIM * HW + h * WDIM + (w0 + m);
  float vals[32];
  float s = 0.0f;
#pragma unroll
  for (int g = 0; g < 4; ++g) {
#pragma unroll
    for (int e = 0; e < 8; ++e) {
      float t = xp[(size_t)(cbase + g * 16 + e) * HW];
      vals[g * 8 + e] = t;
      s += t * t;
    }
  }
  // lane L and L^16 hold disjoint halves of the same row's channels
  s += __shfl_xor(s, 16, 32);
  const float inv = 1.0f / fmaxf(sqrtf(s), 1e-12f);

  v16h a0, a1;
#pragma unroll
  for (int e = 0; e < 16; ++e) {
    a0[e] = (_Float16)(vals[e] * inv);        // channels 0..31 chunk
    a1[e] = (_Float16)(vals[16 + e] * inv);   // channels 32..63 chunk
  }

  // ---- GEMM + running argmax over 32 code tiles.
  float bv[8];
  int   bi[8];
#pragma unroll
  for (int j = 0; j < 8; ++j) { bv[j] = -3.0e38f; bi[j] = 0; }

  for (int T = 0; T < NTILES; ++T) {
    const _Float16* bp = bmat + ((size_t)(T * 2) * 32 + lane) * 16;
    v16h b0 = *(const v16h*)bp;              // K chunk 0 (c 0..31)
    v16h b1 = *(const v16h*)(bp + 32 * 16);  // K chunk 1 (c 32..63)
    v8f acc = {};
    acc = __builtin_amdgcn_wmma_f32_16x16x32_f16(
        false, a0, false, b0, (short)0, acc, false, false);
    acc = __builtin_amdgcn_wmma_f32_16x16x32_f16(
        false, a1, false, b1, (short)0, acc, false, false);
    // D layout: VGPR j, lanes 0-15 -> row j, lanes 16-31 -> row j+8;
    // code index = T*16 + (lane&15). Within-lane idx ascending -> strict '>'
    // keeps the first (smallest) index on ties, matching jnp.argmin.
    const int idx = T * 16 + m;
#pragma unroll
    for (int j = 0; j < 8; ++j) {
      if (acc[j] > bv[j]) { bv[j] = acc[j]; bi[j] = idx; }
    }
  }

  // ---- Reduce across the 16 lanes of each half (rows j / j+8).
#pragma unroll
  for (int j = 0; j < 8; ++j) {
    float v = bv[j];
    int   i = bi[j];
#pragma unroll
    for (int mask = 1; mask <= 8; mask <<= 1) {
      float ov = __shfl_xor(v, mask, 32);
      int   oi = __shfl_xor(i, mask, 32);
      if (ov > v || (ov == v && oi < i)) { v = ov; i = oi; }
    }
    bv[j] = v;
    bi[j] = i;
  }

  // Distribute: lane L needs the winner for row m=L&15. Row m lives in
  // half (m>>3) at slot j=m&7; pick bi[lane&7] then shuffle from the lane
  // of the correct half.
  int sel = bi[0];
#pragma unroll
  for (int j = 1; j < 8; ++j)
    if ((lane & 7) == j) sel = bi[j];
  const int src   = (m & 7) | ((m & 8) << 1);  // lane holding row m's winner
  const int myidx = __shfl(sel, src, 32);

  // ---- indices out: [b,h,w], contiguous over the 16 w's of this wave.
  if (lane < 16)
    idx_out[(size_t)b * HW + h * WDIM + w0 + lane] = (float)myidx;

  // ---- codes out: codes[b,c,h,w0+m] = cbn[myidx][c].
  // Lane covers (m = lane&15, c = 2*it + (lane>>4)); 16-lane-contiguous stores.
  const size_t obase = (size_t)b * CDIM * HW + h * WDIM + (w0 + m);
  const float* crow  = cbn + (size_t)myidx * CDIM;
  const int    chalf = lane >> 4;
#pragma unroll
  for (int it = 0; it < 32; ++it) {
    int c = it * 2 + chalf;
    codes_out[obase + (size_t)c * HW] = crow[c];
  }
}

extern "C" void kernel_launch(void* const* d_in, const int* in_sizes, int n_in,
                              void* d_out, int out_size, void* d_ws,
                              size_t ws_size, hipStream_t stream) {
  (void)in_sizes; (void)n_in; (void)out_size; (void)ws_size;
  const float* x  = (const float*)d_in[0];   // [32,64,64,64]
  const float* cb = (const float*)d_in[1];   // [512,64]

  _Float16* bmat = (_Float16*)d_ws;                          // 64 KB
  float*    cbn  = (float*)((char*)d_ws + WS_BMAT_BYTES);    // 128 KB

  float* codes_out = (float*)d_out;                          // 8388608 f32
  float* idx_out   = codes_out + (size_t)BDIM * CDIM * HW;   // 131072 f32

  vq_prep_kernel<<<1, KCODES, 0, stream>>>(cb, cbn, bmat);

  // 8192 waves total: 1024 blocks x 8 waves (256 threads), exact cover.
  vq_main_kernel<<<1024, 256, 0, stream>>>(x, cbn, bmat, codes_out, idx_out);
}
